// LOE_9320079032336
// MI455X (gfx1250) — compile-verified, matrix operand order checked by reading, not run on hardware
//
#include <hip/hip_runtime.h>

typedef __attribute__((ext_vector_type(8))) int v8i;

#define NS     2500          // 50*50 resized samples per channel
#define NSP    2560          // padded to multiple of 64 (40 j-chunks, 160 i-tiles)
#define NCH    12            // B*C = 4*3
#define IMH    512
#define IMW    512
#define DEN    18750000.0f   // 3 * 2500 * 2500

// ---------------------------------------------------------------------------
// Kernel 1: bilinear resize 512x512 -> 50x50 for both tensors (half_pixel
// coords, matching the reference's two-stage lerp), plus zero the counters.
// ---------------------------------------------------------------------------
__global__ void resize_kernel(const float* __restrict__ in,
                              const float* __restrict__ tg,
                              float* __restrict__ si,
                              float* __restrict__ st,
                              unsigned* __restrict__ cnt) {
    int idx = blockIdx.x * blockDim.x + threadIdx.x;
    if (idx < 4) cnt[idx] = 0u;                 // zero per-batch counters
    if (idx >= NCH * NS) return;

    int ch = idx / NS;                          // b*3 + c
    int p  = idx - ch * NS;
    int oy = p / 50, ox = p - oy * 50;

    const float scale = (float)IMH / 50.0f;     // 10.24
    float sy = fmaxf(((float)oy + 0.5f) * scale - 0.5f, 0.0f);
    int   y0 = min((int)floorf(sy), IMH - 1);
    int   y1 = min(y0 + 1, IMH - 1);
    float wy = sy - (float)y0;

    float sx = fmaxf(((float)ox + 0.5f) * scale - 0.5f, 0.0f);
    int   x0 = min((int)floorf(sx), IMW - 1);
    int   x1 = min(x0 + 1, IMW - 1);
    float wx = sx - (float)x0;

    size_t base = (size_t)ch * IMH * IMW;
    {
        const float* s = in + base;
        float r0 = s[y0 * IMW + x0] * (1.0f - wy) + s[y1 * IMW + x0] * wy;
        float r1 = s[y0 * IMW + x1] * (1.0f - wy) + s[y1 * IMW + x1] * wy;
        si[idx] = r0 * (1.0f - wx) + r1 * wx;
    }
    {
        const float* s = tg + base;
        float r0 = s[y0 * IMW + x0] * (1.0f - wy) + s[y1 * IMW + x0] * wy;
        float r1 = s[y0 * IMW + x1] * (1.0f - wy) + s[y1 * IMW + x1] * wy;
        st[idx] = r0 * (1.0f - wx) + r1 * wx;
    }
}

// ---------------------------------------------------------------------------
// Kernel 2: per (b,c) count discordant pairs using v_wmma_i32_16x16x64_iu8.
// Both LDS arrays are padded with +INF so that every out-of-range row/column
// pair yields ua==ub (rd=0) with NO bounds checks: the inner loop is fully
// branchless, EXEC stays all-ones, and LDS is read with ds_load_b128.
// D[m][n] == rowsum_m for every n (B = all-ones), so
// sum(all 256 D elements) == 16 * true pair count (layout-invariant).
// ---------------------------------------------------------------------------
__global__ __launch_bounds__(256)
void pair_wmma_kernel(const float* __restrict__ si,
                      const float* __restrict__ st,
                      unsigned* __restrict__ cnt) {
    __shared__ float s_si[NSP];
    __shared__ float s_st[NSP];

    const int ch = blockIdx.x;                  // 0..11
    const int b  = ch / 3;
    const float* gi = si + ch * NS;
    const float* gt = st + ch * NS;
    for (int k = threadIdx.x; k < NS; k += blockDim.x) {
        s_si[k] = gi[k];
        s_st[k] = gt[k];
    }
    const float PADV = __builtin_inff();
    for (int k = NS + threadIdx.x; k < NSP; k += blockDim.x) {
        s_si[k] = PADV;                         // pad: rd(pad, x) == 0 always
        s_st[k] = PADV;
    }
    __syncthreads();

    const int lane   = threadIdx.x & 31;
    const int wave   = threadIdx.x >> 5;
    const int row16  = lane & 15;               // A-matrix row = lane % 16
    const int half   = lane >> 4;               // which 32-K half this lane packs
    const int wavesPerBlock = blockDim.x >> 5;
    const int nwaves = wavesPerBlock * gridDim.y;
    const int wid    = blockIdx.y * wavesPerBlock + wave;

    v8i bmat, acc;
#pragma unroll
    for (int k = 0; k < 8; ++k) { bmat[k] = 0x01010101; acc[k] = 0; }

    const int ntiles = NSP / 16;                // 160 i-tiles, all full
    for (int tile = wid; tile < ntiles; tile += nwaves) {
        const int   i  = tile * 16 + row16;     // always in [0, NSP)
        const float sr = s_si[i];
        const float tr = s_st[i];

        for (int j0 = 0; j0 < NSP; j0 += 64) {  // 40 full chunks
            const int jb = j0 + half * 32;      // 128-byte aligned
            const float4* psi = reinterpret_cast<const float4*>(s_si + jb);
            const float4* pst = reinterpret_cast<const float4*>(s_st + jb);
            v8i amat;
#pragma unroll
            for (int r = 0; r < 8; ++r) {
                const float4 a = psi[r];        // ds_load_b128
                const float4 t = pst[r];        // ds_load_b128
                unsigned w = 0;
                w |= ((sr >= a.x) != (tr >= t.x)) ? 0x00000001u : 0u;
                w |= ((sr >= a.y) != (tr >= t.y)) ? 0x00000100u : 0u;
                w |= ((sr >= a.z) != (tr >= t.z)) ? 0x00010000u : 0u;
                w |= ((sr >= a.w) != (tr >= t.w)) ? 0x01000000u : 0u;
                amat[r] = (int)w;
            }
            // D = A(16x64 u8) * ones(64x16 u8) + C   (unsigned, no clamp)
            acc = __builtin_amdgcn_wmma_i32_16x16x64_iu8(
                false, amat, false, bmat, acc, false, false);
        }
    }

    // Reduce: 8 accumulator dwords per lane, then across the 32 lanes.
    int tot = 0;
#pragma unroll
    for (int k = 0; k < 8; ++k) tot += acc[k];
#pragma unroll
    for (int off = 16; off > 0; off >>= 1)
        tot += __shfl_xor(tot, off, 32);

    if (lane == 0 && tot > 0)
        atomicAdd(&cnt[b], (unsigned)(tot >> 4));   // D replicates rowsums x16
}

// ---------------------------------------------------------------------------
// Kernel 3: normalize per-batch counts to the mean.
// ---------------------------------------------------------------------------
__global__ void finalize_kernel(const unsigned* __restrict__ cnt,
                                float* __restrict__ out) {
    int bidx = threadIdx.x;
    if (bidx < 4) out[bidx] = (float)cnt[bidx] / DEN;
}

// ---------------------------------------------------------------------------
extern "C" void kernel_launch(void* const* d_in, const int* in_sizes, int n_in,
                              void* d_out, int out_size, void* d_ws, size_t ws_size,
                              hipStream_t stream) {
    const float* in = (const float*)d_in[0];   // input_tensor  (4,3,512,512) f32
    const float* tg = (const float*)d_in[1];   // target_tensor (4,3,512,512) f32

    float*    si  = (float*)d_ws;              // [12][2500]
    float*    st  = si + NCH * NS;             // [12][2500]
    unsigned* cnt = (unsigned*)(st + NCH * NS);// [4] counters

    const int total = NCH * NS;                // 30000 resize threads
    resize_kernel<<<(total + 255) / 256, 256, 0, stream>>>(in, tg, si, st, cnt);

    dim3 grid(NCH, 8, 1);                      // 12 channels x 8 blocks, 8 waves each
    pair_wmma_kernel<<<grid, 256, 0, stream>>>(si, st, cnt);

    finalize_kernel<<<1, 4, 0, stream>>>(cnt, (float*)d_out);
}
